// MultiHeadAttention_33655363731867
// MI455X (gfx1250) — matrix operand order repaired and checked by reading, not compiled, not run on hardware
//
#include <hip/hip_runtime.h>

// ---------------------------------------------------------------------------
// MI455X (gfx1250) multi-head causal attention, bf16 WMMA path, v2.
// Pre-converted bf16 operands + fragment-order packed weights so the GEMM
// inner loops are pure {global_load_b128 -> v_wmma} streams.
// B=4, S=2048, E=1024, H=16, D=64.  wave32, one wave per GEMM workgroup.
// ---------------------------------------------------------------------------

typedef __attribute__((ext_vector_type(16))) __bf16 v16bf;
typedef __attribute__((ext_vector_type(8)))  __bf16 v8bf;
typedef __attribute__((ext_vector_type(8)))  float  v8f;

#define S_LEN 2048
#define E_DIM 1024
#define H_NUM 16
#define D_DIM 64
#define B_NUM 4
#define NEG_BIG (-3.0e38f)

__device__ __forceinline__ v8f wmma_bf16(v16bf a, v16bf b, v8f c) {
  return __builtin_amdgcn_wmma_f32_16x16x32_bf16(
      false, a, false, b, (short)0, c, false, false);
}

__device__ __forceinline__ v16bf cat8(v8bf lo, v8bf hi) {
  return __builtin_shufflevector(lo, hi, 0, 1, 2, 3, 4, 5, 6, 7,
                                 8, 9, 10, 11, 12, 13, 14, 15);
}

__device__ __forceinline__ v8f v8f_zero() {
  v8f z = {0.f, 0.f, 0.f, 0.f, 0.f, 0.f, 0.f, 0.f};
  return z;
}

// A-fragment (16x32 MxK) from bf16 row-major, element (m,k) at base[m*ld+k].
// Contiguous k order used here (same as B): lane l holds M=l&15; element j holds K=(l>>4)*16 + j
// -> two contiguous 8-element (16 B) runs per lane.
__device__ __forceinline__ v16bf load_a_bf16(const __bf16* __restrict__ base, int ld) {
  const int l = threadIdx.x & 31;
  const __bf16* p = base + (size_t)(l & 15) * ld + ((l >> 4) << 4);
  v8bf lo = *(const v8bf*)(p);
  v8bf hi = *(const v8bf*)(p + 8);
  return cat8(lo, hi);
}

// B-fragment (32x16 KxN) where the K dimension is CONTIGUOUS in memory:
// element (k,n) at base[n*nstride + k].  Contiguous k order used here: lane l holds N=l&15,
// element j holds K=(l>>4)*16 + j -> one 32-byte contiguous run per lane.
__device__ __forceinline__ v16bf load_bk_bf16(const __bf16* __restrict__ base, int nstride) {
  const int l = threadIdx.x & 31;
  const __bf16* p = base + (size_t)(l & 15) * nstride + ((l >> 4) << 4);
  return *(const v16bf*)p;
}

// B-fragment from fragment-order packed weights: lane l's 16 elements are at
// tile + l*16 (32 contiguous bytes).
__device__ __forceinline__ v16bf load_b_packed(const __bf16* __restrict__ tile) {
  return *(const v16bf*)(tile + (size_t)(threadIdx.x & 31) * 16);
}

// ---------------------------------------------------------------------------
// Prep kernels: one-time f32 -> bf16 conversion / fragment-order packing.
// ---------------------------------------------------------------------------
__global__ void __launch_bounds__(256) cvt_x_kernel(const float* __restrict__ X,
                                                    __bf16* __restrict__ Xb) {
  const size_t i = (size_t)blockIdx.x * 256 + threadIdx.x;
  Xb[i] = (__bf16)X[i];
}

// Pack per-head weights (H,E,D): tile order ((h*4 + nt)*32 + kt), each tile
// is 32 lanes x 16 bf16 in WMMA B-fragment order.
__global__ void __launch_bounds__(256) pack_whead_kernel(const float* __restrict__ W,
                                                         __bf16* __restrict__ P) {
  const int idx  = blockIdx.x * 256 + threadIdx.x;
  const int j    = idx & 15;
  const int lane = (idx >> 4) & 31;
  const int tile = idx >> 9;
  const int kt   = tile & 31;        // E/32 = 32 k-tiles
  const int nt   = (tile >> 5) & 3;  // D/16 = 4 n-tiles
  const int h    = tile >> 7;
  const int k    = kt * 32 + ((lane >> 4) << 4) + j;
  const int d    = nt * 16 + (lane & 15);
  P[idx] = (__bf16)W[(size_t)h * E_DIM * D_DIM + (size_t)k * D_DIM + d];
}

// Pack Wo (E x E, row-major): tile order (nt*32 + kt), nt in [0,64).
__global__ void __launch_bounds__(256) pack_wo_kernel(const float* __restrict__ W,
                                                      __bf16* __restrict__ P) {
  const int idx  = blockIdx.x * 256 + threadIdx.x;
  const int j    = idx & 15;
  const int lane = (idx >> 4) & 31;
  const int tile = idx >> 9;
  const int kt   = tile & 31;
  const int nt   = tile >> 5;
  const int k    = kt * 32 + ((lane >> 4) << 4) + j;
  const int n    = nt * 16 + (lane & 15);
  P[idx] = (__bf16)W[(size_t)k * E_DIM + n];
}

// ---------------------------------------------------------------------------
// Kernel 1: fused Q/K/V projection, 32x64 tile per wave.
// grid = 3 * H * (B*S/32) = 12288.  Q,K -> (B,H,S,D) bf16;  V -> (B,H,D,S) bf16.
// ---------------------------------------------------------------------------
__global__ void __launch_bounds__(32) mha_qkv_kernel(
    const __bf16* __restrict__ Xb,
    const __bf16* __restrict__ Pq, const __bf16* __restrict__ Pk, const __bf16* __restrict__ Pv,
    const float* __restrict__ bq, const float* __restrict__ bk, const float* __restrict__ bv,
    __bf16* __restrict__ Q, __bf16* __restrict__ K, __bf16* __restrict__ Vt) {
  const int bid   = blockIdx.x;
  const int mt    = bid & 255;          // 256 M-tiles of 32 rows over B*S
  const int h     = (bid >> 8) & 15;
  const int which = bid >> 12;          // 0=q 1=k 2=v
  const __bf16* Bp   = (which == 0) ? Pq : (which == 1) ? Pk : Pv;
  const float*  bias = (which == 0) ? bq : (which == 1) ? bk : bv;

  const __bf16* A0 = Xb + (size_t)(mt * 32) * E_DIM;
  const __bf16* Bh = Bp + (size_t)h * 4 * 32 * 512;   // per-head packed block

  v8f acc[2][4];
#pragma unroll
  for (int i = 0; i < 2; ++i)
#pragma unroll
    for (int j = 0; j < 4; ++j) acc[i][j] = v8f_zero();

  for (int kt = 0; kt < 32; ++kt) {
    const v16bf a0 = load_a_bf16(A0 + kt * 32, E_DIM);
    const v16bf a1 = load_a_bf16(A0 + (size_t)16 * E_DIM + kt * 32, E_DIM);
#pragma unroll
    for (int nt = 0; nt < 4; ++nt) {
      const v16bf b = load_b_packed(Bh + ((size_t)nt * 32 + kt) * 512);
      acc[0][nt] = wmma_bf16(a0, b, acc[0][nt]);
      acc[1][nt] = wmma_bf16(a1, b, acc[1][nt]);
    }
  }

  const int l = threadIdx.x & 31;
  const int n = l & 15;
  const int half8 = (l >> 4) << 3;
#pragma unroll
  for (int i = 0; i < 2; ++i) {
#pragma unroll
    for (int nt = 0; nt < 4; ++nt) {
      const int d = nt * 16 + n;
      const float bb = bias[h * D_DIM + d];
#pragma unroll
      for (int r = 0; r < 8; ++r) {
        const int mg = mt * 32 + i * 16 + half8 + r;   // row in flattened B*S
        const int bz = mg >> 11;
        const int s  = mg & (S_LEN - 1);
        const __bf16 val = (__bf16)(acc[i][nt][r] + bb);
        if (which == 0)
          Q[(((size_t)(bz * H_NUM + h) * S_LEN + s) * D_DIM) + d] = val;
        else if (which == 1)
          K[(((size_t)(bz * H_NUM + h) * S_LEN + s) * D_DIM) + d] = val;
        else
          Vt[(((size_t)(bz * H_NUM + h) * D_DIM + d) * S_LEN) + s] = val;
      }
    }
  }
}

// ---------------------------------------------------------------------------
// Kernel 2: flash-style causal attention.  grid = B*H*(S/16) = 8192 waves.
// All operand fragments are contiguous bf16 b128 loads.  Writes concat bf16.
// ---------------------------------------------------------------------------
__global__ void __launch_bounds__(32) mha_attn_kernel(
    const __bf16* __restrict__ Q, const __bf16* __restrict__ K,
    const __bf16* __restrict__ Vt, __bf16* __restrict__ Cc) {
  __shared__ __attribute__((aligned(16))) __bf16 Pb[16][40];  // 80B pitch: aligned + conflict-free

  const int bid = blockIdx.x;
  const int qt  = bid & 127;
  const int h   = (bid >> 7) & 15;
  const int bz  = bid >> 11;
  const int q0  = qt * 16;

  const size_t head_off = (size_t)(bz * H_NUM + h) * S_LEN * D_DIM;
  const __bf16* qb = Q  + head_off + (size_t)q0 * D_DIM;
  const __bf16* kb = K  + head_off;
  const __bf16* vb = Vt + head_off;                   // (D, S) per head

  const v16bf qa0 = load_a_bf16(qb + 0,  D_DIM);
  const v16bf qa1 = load_a_bf16(qb + 32, D_DIM);

  const int l = threadIdx.x & 31;
  const int n = l & 15;
  const int m = l & 15;
  const int half8 = (l >> 4) << 3;

  float rm[8], rs[8];
  v8f o[4];
#pragma unroll
  for (int r = 0; r < 8; ++r) { rm[r] = NEG_BIG; rs[r] = 0.f; }
#pragma unroll
  for (int j = 0; j < 4; ++j) o[j] = v8f_zero();

  const float scale = 0.125f;                       // 1/sqrt(64)
  const int nblocks = (q0 + 16 + 31) >> 5;          // 32-key blocks up to diagonal

  for (int t = 0; t < nblocks; ++t) {
    const int kk0 = t * 32;

    if (t + 1 < nblocks) {                          // stream next block into L2/L0
      __builtin_prefetch(kb + (size_t)(kk0 + 32) * D_DIM, 0, 0);
      __builtin_prefetch(vb + (kk0 + 32), 0, 0);
    }

    // ---- scores: 16x32 tile of Q.K^T (4 WMMAs) ----
    v8f s0 = v8f_zero(), s1 = v8f_zero();
    s0 = wmma_bf16(qa0, load_bk_bf16(kb + (size_t)kk0 * D_DIM + 0,  D_DIM), s0);
    s0 = wmma_bf16(qa1, load_bk_bf16(kb + (size_t)kk0 * D_DIM + 32, D_DIM), s0);
    s1 = wmma_bf16(qa0, load_bk_bf16(kb + (size_t)(kk0 + 16) * D_DIM + 0,  D_DIM), s1);
    s1 = wmma_bf16(qa1, load_bk_bf16(kb + (size_t)(kk0 + 16) * D_DIM + 32, D_DIM), s1);

    // ---- scale + causal mask + online softmax ----
    float nm[8];
#pragma unroll
    for (int r = 0; r < 8; ++r) {
      const int row = q0 + half8 + r;
      float x0 = s0[r] * scale; if (kk0 + n      > row) x0 = NEG_BIG;
      float x1 = s1[r] * scale; if (kk0 + 16 + n > row) x1 = NEG_BIG;
      s0[r] = x0; s1[r] = x1;
      float mx = fmaxf(x0, x1);
      mx = fmaxf(mx, __shfl_xor(mx, 1, 32));
      mx = fmaxf(mx, __shfl_xor(mx, 2, 32));
      mx = fmaxf(mx, __shfl_xor(mx, 4, 32));
      mx = fmaxf(mx, __shfl_xor(mx, 8, 32));
      nm[r] = fmaxf(rm[r], mx);
    }
#pragma unroll
    for (int r = 0; r < 8; ++r) {
      const float alpha = __expf(rm[r] - nm[r]);
      const float p0 = __expf(s0[r] - nm[r]);
      const float p1 = __expf(s1[r] - nm[r]);
      float ps = p0 + p1;
      ps += __shfl_xor(ps, 1, 32);
      ps += __shfl_xor(ps, 2, 32);
      ps += __shfl_xor(ps, 4, 32);
      ps += __shfl_xor(ps, 8, 32);
      rs[r] = rs[r] * alpha + ps;
      rm[r] = nm[r];
      s0[r] = p0; s1[r] = p1;
#pragma unroll
      for (int j = 0; j < 4; ++j) o[j][r] *= alpha;
    }

    // ---- P: C-layout -> A-layout via LDS (bf16, 128-bit reloads) ----
    __syncthreads();
#pragma unroll
    for (int r = 0; r < 8; ++r) {
      Pb[half8 + r][n]      = (__bf16)s0[r];
      Pb[half8 + r][16 + n] = (__bf16)s1[r];
    }
    __syncthreads();
    const __bf16* pp = &Pb[m][(l >> 4) << 4];
    const v16bf pa = cat8(*(const v8bf*)pp, *(const v8bf*)(pp + 8));

    // ---- O += P.V  (V^T layout: key dim contiguous -> b128 B-frags) ----
#pragma unroll
    for (int j = 0; j < 4; ++j) {
      const v16bf vf = load_bk_bf16(vb + (size_t)(j * 16) * S_LEN + kk0, S_LEN);
      o[j] = wmma_bf16(pa, vf, o[j]);
    }
  }

  // ---- normalize, store concat (B, S, H*D) bf16 ----
#pragma unroll
  for (int r = 0; r < 8; ++r) {
    const float inv = 1.0f / rs[r];
    const int s = q0 + half8 + r;
#pragma unroll
    for (int j = 0; j < 4; ++j) {
      const int d = j * 16 + n;
      Cc[((size_t)bz * S_LEN + s) * E_DIM + h * D_DIM + d] = (__bf16)(o[j][r] * inv);
    }
  }
}

// ---------------------------------------------------------------------------
// Kernel 3: output projection Y = concat @ Wo + bo, 32x64 tile per wave.
// grid = (B*S/32) * (E/64) = 4096.
// ---------------------------------------------------------------------------
__global__ void __launch_bounds__(32) mha_out_kernel(
    const __bf16* __restrict__ Cc, const __bf16* __restrict__ Po,
    const float* __restrict__ bo, float* __restrict__ Y) {
  const int bid = blockIdx.x;
  const int mt  = bid & 255;
  const int ntB = bid >> 8;            // 16 blocks of 64 output columns

  const __bf16* A0 = Cc + (size_t)(mt * 32) * E_DIM;

  v8f acc[2][4];
#pragma unroll
  for (int i = 0; i < 2; ++i)
#pragma unroll
    for (int j = 0; j < 4; ++j) acc[i][j] = v8f_zero();

  for (int kt = 0; kt < 32; ++kt) {
    const v16bf a0 = load_a_bf16(A0 + kt * 32, E_DIM);
    const v16bf a1 = load_a_bf16(A0 + (size_t)16 * E_DIM + kt * 32, E_DIM);
#pragma unroll
    for (int j = 0; j < 4; ++j) {
      const v16bf b = load_b_packed(Po + ((size_t)(ntB * 4 + j) * 32 + kt) * 512);
      acc[0][j] = wmma_bf16(a0, b, acc[0][j]);
      acc[1][j] = wmma_bf16(a1, b, acc[1][j]);
    }
  }

  const int l = threadIdx.x & 31;
  const int n = l & 15;
  const int half8 = (l >> 4) << 3;
#pragma unroll
  for (int i = 0; i < 2; ++i) {
#pragma unroll
    for (int j = 0; j < 4; ++j) {
      const int e = ntB * 64 + j * 16 + n;
      const float bb = bo[e];
#pragma unroll
      for (int r = 0; r < 8; ++r) {
        const int mg = mt * 32 + i * 16 + half8 + r;
        Y[(size_t)mg * E_DIM + e] = acc[i][j][r] + bb;
      }
    }
  }
}

// ---------------------------------------------------------------------------
extern "C" void kernel_launch(void* const* d_in, const int* in_sizes, int n_in,
                              void* d_out, int out_size, void* d_ws, size_t ws_size,
                              hipStream_t stream) {
  const float* X  = (const float*)d_in[0];
  const float* Wq = (const float*)d_in[1];
  const float* Wk = (const float*)d_in[2];
  const float* Wv = (const float*)d_in[3];
  const float* bq = (const float*)d_in[4];
  const float* bk = (const float*)d_in[5];
  const float* bv = (const float*)d_in[6];
  const float* Wo = (const float*)d_in[7];
  const float* bo = (const float*)d_in[8];
  float* Y = (float*)d_out;

  // bf16 workspace: Xb,Qb,Kb,Vt,Cc (8.4M elems each) + 4 packed weights (1M each) ~ 88 MB
  const size_t big = (size_t)B_NUM * S_LEN * E_DIM;   // == B*H*S*D == 8388608
  const size_t wsz = (size_t)E_DIM * E_DIM;           // 1048576
  __bf16* Xb = (__bf16*)d_ws;
  __bf16* Qb = Xb + big;
  __bf16* Kb = Qb + big;
  __bf16* Vt = Kb + big;
  __bf16* Cc = Vt + big;
  __bf16* Pq = Cc + big;
  __bf16* Pk = Pq + wsz;
  __bf16* Pv = Pk + wsz;
  __bf16* Po = Pv + wsz;

  cvt_x_kernel<<<(int)(big / 256), 256, 0, stream>>>(X, Xb);
  pack_whead_kernel<<<(int)(wsz / 256), 256, 0, stream>>>(Wq, Pq);
  pack_whead_kernel<<<(int)(wsz / 256), 256, 0, stream>>>(Wk, Pk);
  pack_whead_kernel<<<(int)(wsz / 256), 256, 0, stream>>>(Wv, Pv);
  pack_wo_kernel<<<(int)(wsz / 256), 256, 0, stream>>>(Wo, Po);

  mha_qkv_kernel<<<12288, 32, 0, stream>>>(Xb, Pq, Pk, Pv, bq, bk, bv, Qb, Kb, Vt);
  mha_attn_kernel<<<8192, 32, 0, stream>>>(Qb, Kb, Vt, Cc);
  mha_out_kernel<<<4096, 32, 0, stream>>>(Cc, Po, bo, Y);
}
